// TurboQuantMSE_49168785605237
// MI455X (gfx1250) — matrix operand order repaired and checked by reading, not compiled, not run on hardware
//
#include <hip/hip_runtime.h>
#include <hip/hip_bf16.h>

#define DIM 4096
#define BM 128
#define BN 128
#define BK 32
#define LP 40          // LDS row pitch in bf16 elements (32 + 8 pad -> conflict-free b128 reads)
#define THREADS 256

typedef __attribute__((ext_vector_type(16))) __bf16 bf16x16;
typedef __attribute__((ext_vector_type(8)))  float  f32x8;

// f32 -> bf16 (round to nearest even, ignoring NaN corner)
static __device__ __forceinline__ unsigned short f2bf(float f) {
    unsigned u = __builtin_bit_cast(unsigned, f);
    u += 0x7FFFu + ((u >> 16) & 1u);
    return (unsigned short)(u >> 16);
}

static __device__ __forceinline__ bf16x16 make_frag(uint4 lo, uint4 hi) {
    union { uint4 u[2]; bf16x16 v; } c;
    c.u[0] = lo; c.u[1] = hi;
    return c.v;
}

// ---------------------------------------------------------------------------
// Kernel 1: y = x @ Pi^T  (bf16 WMMA, f32 acc), fused 16-level quantization,
// emit y_tilde as bf16 row-major [DIM][DIM] into workspace.
// ---------------------------------------------------------------------------
__global__ __launch_bounds__(THREADS)
void gemm1_quant(const float* __restrict__ x, const float* __restrict__ Pi,
                 const float* __restrict__ cb, unsigned short* __restrict__ yt)
{
    __shared__ unsigned short lA[BM * LP];
    __shared__ unsigned short lB[BN * LP];

    const int tid   = threadIdx.x;
    const int lane  = tid & 31;
    const int wid   = tid >> 5;
    const int waveM = (wid >> 2) * 64;   // 0 or 64
    const int waveN = (wid & 3) * 32;    // 0,32,64,96
    const int mBase = blockIdx.y * BM;
    const int nBase = blockIdx.x * BN;
    const int lm    = lane & 15;
    const int hiL   = lane >> 4;

    f32x8 acc[4][2];
    #pragma unroll
    for (int i = 0; i < 4; ++i)
        #pragma unroll
        for (int j = 0; j < 2; ++j)
            #pragma unroll
            for (int e = 0; e < 8; ++e) acc[i][j][e] = 0.0f;

    for (int k0 = 0; k0 < DIM; k0 += BK) {
        // Stage A tile: x[mBase..+127][k0..+31], f32 -> bf16
        #pragma unroll
        for (int it = 0; it < 4; ++it) {
            int idx = tid + it * THREADS;            // 0..1023 float4 slots
            int row = idx >> 3, q = idx & 7;
            float4 f = reinterpret_cast<const float4*>(x + (size_t)(mBase + row) * DIM + k0)[q];
            ushort4 h; h.x = f2bf(f.x); h.y = f2bf(f.y); h.z = f2bf(f.z); h.w = f2bf(f.w);
            *reinterpret_cast<ushort4*>(&lA[row * LP + q * 4]) = h;
        }
        // Stage B tile: Bt[n][k] = Pi[nBase+n][k0+k] (NT: contiguous along K)
        #pragma unroll
        for (int it = 0; it < 4; ++it) {
            int idx = tid + it * THREADS;
            int row = idx >> 3, q = idx & 7;
            float4 f = reinterpret_cast<const float4*>(Pi + (size_t)(nBase + row) * DIM + k0)[q];
            ushort4 h; h.x = f2bf(f.x); h.y = f2bf(f.y); h.z = f2bf(f.z); h.w = f2bf(f.w);
            *reinterpret_cast<ushort4*>(&lB[row * LP + q * 4]) = h;
        }
        __syncthreads();

        bf16x16 af[4], bfr[2];
        #pragma unroll
        for (int mt = 0; mt < 4; ++mt) {
            const unsigned short* p = &lA[(waveM + mt * 16 + lm) * LP + hiL * 8];
            af[mt] = make_frag(*reinterpret_cast<const uint4*>(p),
                               *reinterpret_cast<const uint4*>(p + 16));
        }
        #pragma unroll
        for (int nt = 0; nt < 2; ++nt) {
            const unsigned short* p = &lB[(waveN + nt * 16 + lm) * LP + hiL * 8];
            bfr[nt] = make_frag(*reinterpret_cast<const uint4*>(p),
                                *reinterpret_cast<const uint4*>(p + 16));
        }
        #pragma unroll
        for (int mt = 0; mt < 4; ++mt)
            #pragma unroll
            for (int nt = 0; nt < 2; ++nt)
                acc[mt][nt] = __builtin_amdgcn_wmma_f32_16x16x32_bf16(
                    false, af[mt], false, bfr[nt], (short)0, acc[mt][nt], false, false);
        __syncthreads();
    }

    // Quantize to nearest of 16 sorted codebook levels (midpoint cndmask chain)
    float c[16];
    #pragma unroll
    for (int i = 0; i < 16; ++i) c[i] = cb[i];

    #pragma unroll
    for (int mt = 0; mt < 4; ++mt)
        #pragma unroll
        for (int nt = 0; nt < 2; ++nt)
            #pragma unroll
            for (int v = 0; v < 8; ++v) {
                float y   = acc[mt][nt][v];
                float val = c[0];
                #pragma unroll
                for (int j = 0; j < 15; ++j) {
                    float mid = 0.5f * (c[j] + c[j + 1]);
                    val = (y > mid) ? c[j + 1] : val;
                }
                int m = mBase + waveM + mt * 16 + v + hiL * 8;
                int n = nBase + waveN + nt * 16 + lm;
                yt[(size_t)m * DIM + n] = f2bf(val);
            }
}

// ---------------------------------------------------------------------------
// Kernel 2: x_tilde = y_tilde @ Pi  (bf16 WMMA, f32 acc), f32 output.
// ---------------------------------------------------------------------------
__global__ __launch_bounds__(THREADS)
void gemm2(const unsigned short* __restrict__ yt, const float* __restrict__ Pi,
           float* __restrict__ out)
{
    __shared__ unsigned short lA[BM * LP];
    __shared__ unsigned short lB[BN * LP];

    const int tid   = threadIdx.x;
    const int lane  = tid & 31;
    const int wid   = tid >> 5;
    const int waveM = (wid >> 2) * 64;
    const int waveN = (wid & 3) * 32;
    const int mBase = blockIdx.y * BM;
    const int nBase = blockIdx.x * BN;
    const int lm    = lane & 15;
    const int hiL   = lane >> 4;

    f32x8 acc[4][2];
    #pragma unroll
    for (int i = 0; i < 4; ++i)
        #pragma unroll
        for (int j = 0; j < 2; ++j)
            #pragma unroll
            for (int e = 0; e < 8; ++e) acc[i][j][e] = 0.0f;

    for (int k0 = 0; k0 < DIM; k0 += BK) {
        // Stage A tile: y_tilde already bf16, straight 16B copies
        #pragma unroll
        for (int it = 0; it < 2; ++it) {
            int idx = tid + it * THREADS;            // 0..511 uint4 slots
            int row = idx >> 2, q = idx & 3;
            uint4 d = reinterpret_cast<const uint4*>(yt + (size_t)(mBase + row) * DIM + k0)[q];
            *reinterpret_cast<uint4*>(&lA[row * LP + q * 8]) = d;
        }
        // Stage B tile (NN): Bt[n][k] = Pi[k0+k][nBase+n] -> transpose on LDS store
        #pragma unroll
        for (int it = 0; it < 4; ++it) {
            int idx = tid + it * THREADS;            // 0..1023 float4 slots
            int kk = idx >> 5, qn = idx & 31;
            float4 f = reinterpret_cast<const float4*>(Pi + (size_t)(k0 + kk) * DIM + nBase)[qn];
            lB[(qn * 4 + 0) * LP + kk] = f2bf(f.x);
            lB[(qn * 4 + 1) * LP + kk] = f2bf(f.y);
            lB[(qn * 4 + 2) * LP + kk] = f2bf(f.z);
            lB[(qn * 4 + 3) * LP + kk] = f2bf(f.w);
        }
        __syncthreads();

        bf16x16 af[4], bfr[2];
        #pragma unroll
        for (int mt = 0; mt < 4; ++mt) {
            const unsigned short* p = &lA[(waveM + mt * 16 + lm) * LP + hiL * 8];
            af[mt] = make_frag(*reinterpret_cast<const uint4*>(p),
                               *reinterpret_cast<const uint4*>(p + 16));
        }
        #pragma unroll
        for (int nt = 0; nt < 2; ++nt) {
            const unsigned short* p = &lB[(waveN + nt * 16 + lm) * LP + hiL * 8];
            bfr[nt] = make_frag(*reinterpret_cast<const uint4*>(p),
                                *reinterpret_cast<const uint4*>(p + 16));
        }
        #pragma unroll
        for (int mt = 0; mt < 4; ++mt)
            #pragma unroll
            for (int nt = 0; nt < 2; ++nt)
                acc[mt][nt] = __builtin_amdgcn_wmma_f32_16x16x32_bf16(
                    false, af[mt], false, bfr[nt], (short)0, acc[mt][nt], false, false);
        __syncthreads();
    }

    #pragma unroll
    for (int mt = 0; mt < 4; ++mt)
        #pragma unroll
        for (int nt = 0; nt < 2; ++nt)
            #pragma unroll
            for (int v = 0; v < 8; ++v) {
                int m = mBase + waveM + mt * 16 + v + hiL * 8;
                int n = nBase + waveN + nt * 16 + lm;
                out[(size_t)m * DIM + n] = acc[mt][nt][v];
            }
}

extern "C" void kernel_launch(void* const* d_in, const int* in_sizes, int n_in,
                              void* d_out, int out_size, void* d_ws, size_t ws_size,
                              hipStream_t stream) {
    (void)in_sizes; (void)n_in; (void)out_size; (void)ws_size;
    const float* x  = (const float*)d_in[0];
    const float* Pi = (const float*)d_in[1];
    const float* cb = (const float*)d_in[2];
    float* out = (float*)d_out;
    unsigned short* yt = (unsigned short*)d_ws;   // 4096*4096 bf16 = 32 MB scratch

    dim3 grid(DIM / BN, DIM / BM);
    dim3 block(THREADS);
    gemm1_quant<<<grid, block, 0, stream>>>(x, Pi, cb, yt);
    gemm2<<<grid, block, 0, stream>>>(yt, Pi, out);
}
